// FeedForward_45037027065973
// MI455X (gfx1250) — matrix operand order repaired and checked by reading, not compiled, not run on hardware
//
#include <hip/hip_runtime.h>
#include <hip/hip_bf16.h>

typedef __attribute__((ext_vector_type(16))) __bf16 v16bf;
typedef __attribute__((ext_vector_type(8)))  float  v8f;

#define NTOK 8192
#define DDIM 1024
#define FDIM 4096
#define NEXP 8
#define TM   32          // tokens per FFN block
#define FC   128         // F-chunk size
#define TILES_PER_E 256  // ceil(NTOK / TM)

#define KT1  (DDIM / 32) // 32  K-tiles for GEMM1
#define FT1  (FDIM / 16) // 256 N-tiles for GEMM1
#define KT2  (FDIM / 32) // 128 K-tiles for GEMM2
#define DT2  (DDIM / 16) // 64  N-tiles for GEMM2

union AFrag { v16bf v; unsigned u[8]; };
union BFrag { v16bf v; __bf16  h[16]; };

// ---------------- Gating: logits -> softmax -> top2 -> scatter by expert ---
__global__ __launch_bounds__(256) void moe_gate(
    const float* __restrict__ x, const float* __restrict__ gw,
    const float* __restrict__ gb, int* __restrict__ counts,
    int* __restrict__ ids, float* __restrict__ wts)
{
    int wave = threadIdx.x >> 5;
    int lane = threadIdx.x & 31;
    int t = blockIdx.x * 8 + wave;           // one wave32 per token

    float acc[NEXP];
#pragma unroll
    for (int e = 0; e < NEXP; ++e) acc[e] = 0.f;

    const float* xr = x + (size_t)t * DDIM;
    for (int d = lane; d < DDIM; d += 32) {
        float xv = xr[d];
        const float* g = gw + (size_t)d * NEXP;
#pragma unroll
        for (int e = 0; e < NEXP; ++e) acc[e] = fmaf(xv, g[e], acc[e]);
    }
#pragma unroll
    for (int e = 0; e < NEXP; ++e)
        for (int off = 16; off > 0; off >>= 1)
            acc[e] += __shfl_xor(acc[e], off, 32);

    if (lane == 0) {
        float lg[NEXP], pr[NEXP];
        float m = -3.0e38f;
#pragma unroll
        for (int e = 0; e < NEXP; ++e) { lg[e] = acc[e] + gb[e]; m = fmaxf(m, lg[e]); }
        float s = 0.f;
#pragma unroll
        for (int e = 0; e < NEXP; ++e) { pr[e] = __expf(lg[e] - m); s += pr[e]; }
        float inv = 1.f / s;
#pragma unroll
        for (int e = 0; e < NEXP; ++e) pr[e] *= inv;

        int i0 = 0; float p0 = pr[0];
#pragma unroll
        for (int e = 1; e < NEXP; ++e) if (pr[e] > p0) { p0 = pr[e]; i0 = e; }
        int i1 = -1; float p1 = -1.f;
#pragma unroll
        for (int e = 0; e < NEXP; ++e) if (e != i0 && pr[e] > p1) { p1 = pr[e]; i1 = e; }

        float rs = 1.f / (p0 + p1);
        float w0 = p0 * rs, w1v = p1 * rs;
        int pos0 = atomicAdd(&counts[i0], 1);
        ids[i0 * NTOK + pos0] = t; wts[i0 * NTOK + pos0] = w0;
        int pos1 = atomicAdd(&counts[i1], 1);
        ids[i1 * NTOK + pos1] = t; wts[i1 * NTOK + pos1] = w1v;
    }
}

// ---- Weight pre-swizzle: fp32 -> bf16 in exact WMMA B-fragment lane order --
// w1s frag-lane id bits: [20:18]=e [17:10]=ft [9:5]=kt [4:0]=lane
__global__ __launch_bounds__(256) void conv_w1(
    const float* __restrict__ w1, __bf16* __restrict__ w1s)
{
    unsigned id = blockIdx.x * 256 + threadIdx.x;
    int l  = id & 31;
    int kt = (id >> 5) & 31;
    int ft = (id >> 10) & 255;
    int e  = id >> 18;
    int n  = l & 15;
    int kb = (l < 16) ? 0 : 8;
    const float* src = w1 + ((size_t)e * DDIM + (size_t)kt * 32 + kb) * FDIM
                          + (size_t)ft * 16 + n;
    BFrag f;
#pragma unroll
    for (int h = 0; h < 16; ++h) f.h[h] = (__bf16)src[(size_t)(h + (h & 8)) * FDIM];
    *(v16bf*)(w1s + (size_t)id * 16) = f.v;
}

// w2s frag-lane id bits: [20:18]=e [17:12]=dt [11:5]=ktf [4:0]=lane
__global__ __launch_bounds__(256) void conv_w2(
    const float* __restrict__ w2, __bf16* __restrict__ w2s)
{
    unsigned id = blockIdx.x * 256 + threadIdx.x;
    int l   = id & 31;
    int ktf = (id >> 5) & 127;
    int dt  = (id >> 12) & 63;
    int e   = id >> 18;
    int n   = l & 15;
    int kb  = (l < 16) ? 0 : 8;
    const float* src = w2 + ((size_t)e * FDIM + (size_t)ktf * 32 + kb) * DDIM
                          + (size_t)dt * 16 + n;
    BFrag f;
#pragma unroll
    for (int h = 0; h < 16; ++h) f.h[h] = (__bf16)src[(size_t)(h + (h & 8)) * DDIM];
    *(v16bf*)(w2s + (size_t)id * 16) = f.v;
}

// ---------------- Fast FFN: swizzled bf16 weights, b128 fragment loads ------
__global__ __launch_bounds__(256) void moe_ffn_pre(
    const float* __restrict__ x,
    const __bf16* __restrict__ w1s, const float* __restrict__ b1,
    const __bf16* __restrict__ w2s, const float* __restrict__ b2,
    const int* __restrict__ counts, const int* __restrict__ ids,
    const float* __restrict__ wts, float* __restrict__ out)
{
    extern __shared__ char smem[];
    __bf16* xs  = (__bf16*)smem;                              // 2*KT1*32 frag-lanes
    __bf16* mid = (__bf16*)(smem + (size_t)TM * DDIM * 2);    // 2*4*32 frag-lanes
    int*   stok = (int*)  (smem + (size_t)TM * DDIM * 2 + (size_t)TM * FC * 2);
    float* swgt = (float*)(stok + TM);

    int e    = blockIdx.x / TILES_PER_E;
    int tile = blockIdx.x % TILES_PER_E;
    int cnt  = counts[e];
    int base = tile * TM;
    if (base >= cnt) return;                 // uniform per block

    int tid  = threadIdx.x;
    int wave = tid >> 5;
    int lane = tid & 31;

    if (tid < TM) {
        int slot = base + tid;
        int tk = 0; float w = 0.f;
        if (slot < cnt) { tk = ids[e * NTOK + slot]; w = wts[e * NTOK + slot]; }
        stok[tid] = tk; swgt[tid] = w;
    }
    __syncthreads();

    // Stage gathered x rows as A-fragments in LDS (bf16, frag-lane major).
    // frag id bits: [10]=mt [9:5]=kt [4:0]=lane ; half h -> K = kb + h + (h&8)
    for (int fid = tid; fid < 2 * KT1 * 32; fid += 256) {
        int l  = fid & 31;
        int kt = (fid >> 5) & 31;
        int mt = fid >> 10;
        int m  = l & 15;
        int kb = (l < 16) ? 0 : 8;
        const float* src = x + (size_t)stok[mt * 16 + m] * DDIM + kt * 32 + kb;
        BFrag f;
#pragma unroll
        for (int h = 0; h < 16; ++h) f.h[h] = (__bf16)src[h + (h & 8)];
        *(v16bf*)(xs + (size_t)fid * 16) = f.v;
    }
    __syncthreads();

    const v16bf* xv  = (const v16bf*)xs;
    const v16bf* mv  = (const v16bf*)mid;
    const v16bf* w1v = (const v16bf*)w1s;
    const v16bf* w2v = (const v16bf*)w2s;
    const float* b1e = b1 + (size_t)e * FDIM;
    const float* b2e = b2 + (size_t)e * DDIM;

    const int nB  = lane & 15;
    const int rsh = (lane < 16) ? 0 : 8;

    v8f acc[2][8];
#pragma unroll
    for (int a = 0; a < 2; ++a)
#pragma unroll
        for (int b_ = 0; b_ < 8; ++b_) {
            v8f z = {0.f,0.f,0.f,0.f,0.f,0.f,0.f,0.f};
            acc[a][b_] = z;
        }

    for (int c = 0; c < FDIM / FC; ++c) {
        int ft = c * 8 + wave;               // this wave's GEMM1 N-tile
        size_t base1 = (((size_t)e * FT1 + ft) * KT1) * 32;

        // Prefetch next chunk's w1 fragment block for this wave (CDNA5 path)
        if (c + 1 < FDIM / FC) {
            const char* pf = (const char*)(w1v + (((size_t)e * FT1 + ft + 8) * KT1) * 32 + lane);
            __builtin_prefetch(pf, 0, 1);
        }

        // ---- GEMM1: mid[0..31][wave*16..+15] = relu(x @ w1 + b1) over K=D
        v8f acc1[2];
#pragma unroll
        for (int mi = 0; mi < 2; ++mi) {
            v8f z = {0.f,0.f,0.f,0.f,0.f,0.f,0.f,0.f};
            acc1[mi] = z;
        }
#pragma unroll 4
        for (int kt = 0; kt < KT1; ++kt) {
            BFrag bf; bf.v = w1v[base1 + (size_t)kt * 32 + lane];
#pragma unroll
            for (int mi = 0; mi < 2; ++mi) {
                AFrag af; af.v = xv[(size_t)(mi * KT1 + kt) * 32 + lane];
                acc1[mi] = __builtin_amdgcn_wmma_f32_16x16x32_bf16(
                    false, af.v, false, bf.v, (short)0, acc1[mi], false, false);
            }
        }
        float b1f = b1e[c * FC + wave * 16 + nB];
        __syncthreads();                      // prior GEMM2 readers of mid done
        {
            // scatter D-regs into mid A-fragment layout
            int l2  = (nB & 8) ? 16 : 0;      // +16 lanes if n>=8
            int h   = (nB & 7) + ((wave & 1) ? 8 : 0);
            int ktm = wave >> 1;
#pragma unroll
            for (int mi = 0; mi < 2; ++mi)
#pragma unroll
                for (int r = 0; r < 8; ++r) {
                    float v = acc1[mi][r] + b1f;
                    v = v > 0.f ? v : 0.f;    // ReLU
                    int mrow = (r + rsh) & 15;
                    mid[((size_t)(mi * 4 + ktm) * 32 + mrow + l2) * 16 + h] = (__bf16)v;
                }
        }
        __syncthreads();

        // ---- GEMM2: out[32][wave*128..+127] += mid @ w2 over this F chunk
#pragma unroll 1
        for (int k2 = 0; k2 < 4; ++k2) {
            int ktf = c * 4 + k2;
            AFrag af[2];
#pragma unroll
            for (int mi = 0; mi < 2; ++mi)
                af[mi].v = mv[(size_t)(mi * 4 + k2) * 32 + lane];
#pragma unroll
            for (int ni = 0; ni < 8; ++ni) {
                int dt = wave * 8 + ni;
                BFrag bf;
                bf.v = w2v[(((size_t)e * DT2 + dt) * KT2 + ktf) * 32 + lane];
#pragma unroll
                for (int mi = 0; mi < 2; ++mi)
                    acc[mi][ni] = __builtin_amdgcn_wmma_f32_16x16x32_bf16(
                        false, af[mi].v, false, bf.v, (short)0, acc[mi][ni], false, false);
            }
        }
    }

    // ---- Epilogue: weighted atomic combine (2 real addends/elem -> deterministic)
#pragma unroll
    for (int mi = 0; mi < 2; ++mi)
#pragma unroll
        for (int ni = 0; ni < 8; ++ni) {
            int dcol = wave * 128 + ni * 16 + nB;
            float bb = b2e[dcol];
#pragma unroll
            for (int r = 0; r < 8; ++r) {
                int row = mi * 16 + r + rsh;
                float contrib = swgt[row] * (acc[mi][ni][r] + bb);
                atomicAdd(&out[(size_t)stok[row] * DDIM + dcol], contrib);
            }
        }
}

// ---------------- Fallback FFN: fp32 weights converted on the fly -----------
__global__ __launch_bounds__(256) void moe_ffn_raw(
    const float* __restrict__ x,
    const float* __restrict__ w1, const float* __restrict__ b1,
    const float* __restrict__ w2, const float* __restrict__ b2,
    const int* __restrict__ counts, const int* __restrict__ ids,
    const float* __restrict__ wts, float* __restrict__ out)
{
    extern __shared__ char smem[];
    __bf16* xs  = (__bf16*)smem;
    __bf16* mid = (__bf16*)(smem + (size_t)TM * DDIM * 2);
    int*   stok = (int*)  (smem + (size_t)TM * DDIM * 2 + (size_t)TM * FC * 2);
    float* swgt = (float*)(stok + TM);

    int e    = blockIdx.x / TILES_PER_E;
    int tile = blockIdx.x % TILES_PER_E;
    int cnt  = counts[e];
    int base = tile * TM;
    if (base >= cnt) return;

    int tid  = threadIdx.x;
    int wave = tid >> 5;
    int lane = tid & 31;

    if (tid < TM) {
        int slot = base + tid;
        int tk = 0; float w = 0.f;
        if (slot < cnt) { tk = ids[e * NTOK + slot]; w = wts[e * NTOK + slot]; }
        stok[tid] = tk; swgt[tid] = w;
    }
    __syncthreads();

    for (int idx = tid; idx < TM * DDIM; idx += 256) {
        int r = idx >> 10, cc = idx & (DDIM - 1);
        xs[idx] = (__bf16)x[(size_t)stok[r] * DDIM + cc];
    }
    __syncthreads();

    const float* w1e = w1 + (size_t)e * DDIM * FDIM;
    const float* w2e = w2 + (size_t)e * FDIM * DDIM;
    const float* b1e = b1 + (size_t)e * FDIM;
    const float* b2e = b2 + (size_t)e * DDIM;

    const int m   = lane & 15;
    const int kbA = (lane < 16) ? 0 : 16;
    const int nB  = lane & 15;
    const int kbB = (lane < 16) ? 0 : 16;
    const int rsh = (lane < 16) ? 0 : 8;

    v8f acc[2][8];
#pragma unroll
    for (int a = 0; a < 2; ++a)
#pragma unroll
        for (int b_ = 0; b_ < 8; ++b_) {
            v8f z = {0.f,0.f,0.f,0.f,0.f,0.f,0.f,0.f};
            acc[a][b_] = z;
        }

    for (int c = 0; c < FDIM / FC; ++c) {
        int fbase = c * FC;
        int fcol  = fbase + wave * 16 + nB;

        v8f acc1[2];
#pragma unroll
        for (int mi = 0; mi < 2; ++mi) {
            v8f z = {0.f,0.f,0.f,0.f,0.f,0.f,0.f,0.f};
            acc1[mi] = z;
        }
        for (int kk = 0; kk < DDIM; kk += 32) {
            BFrag bf;
#pragma unroll
            for (int j = 0; j < 8; ++j) {
                const float* p = w1e + (size_t)(kk + kbB + 2 * j) * FDIM + fcol;
                bf.h[2 * j]     = (__bf16)p[0];
                bf.h[2 * j + 1] = (__bf16)p[FDIM];
            }
#pragma unroll
            for (int mi = 0; mi < 2; ++mi) {
                AFrag af;
                const unsigned* rp = (const unsigned*)(xs + (size_t)(mi * 16 + m) * DDIM);
                int b0 = (kk + kbA) >> 1;
                int b1i = (kk + 8 + kbA) >> 1;
#pragma unroll
                for (int j = 0; j < 4; ++j) { af.u[j] = rp[b0 + j]; af.u[4 + j] = rp[b1i + j]; }
                acc1[mi] = __builtin_amdgcn_wmma_f32_16x16x32_bf16(
                    false, af.v, false, bf.v, (short)0, acc1[mi], false, false);
            }
        }
        float b1f = b1e[fcol];
        __syncthreads();
#pragma unroll
        for (int mi = 0; mi < 2; ++mi)
#pragma unroll
            for (int r = 0; r < 8; ++r) {
                float v = acc1[mi][r] + b1f;
                v = v > 0.f ? v : 0.f;
                int row = mi * 16 + r + rsh;
                mid[row * FC + wave * 16 + nB] = (__bf16)v;
            }
        __syncthreads();

        int dbase = wave * 128;
#pragma unroll 1
        for (int kk = 0; kk < FC; kk += 32) {
            AFrag af[2];
#pragma unroll
            for (int mi = 0; mi < 2; ++mi) {
                const unsigned* rp = (const unsigned*)(mid + (size_t)(mi * 16 + m) * FC);
                int b0 = (kk + kbA) >> 1;
                int b1i = (kk + 8 + kbA) >> 1;
#pragma unroll
                for (int j = 0; j < 4; ++j) { af[mi].u[j] = rp[b0 + j]; af[mi].u[4 + j] = rp[b1i + j]; }
            }
#pragma unroll
            for (int ni = 0; ni < 8; ++ni) {
                int dcol = dbase + ni * 16 + nB;
                const float* p2 = w2e + (size_t)(fbase + kk + kbB) * DDIM + dcol;
                BFrag bf;
#pragma unroll
                for (int j = 0; j < 8; ++j) {
                    bf.h[2 * j]     = (__bf16)p2[(size_t)(2 * j) * DDIM];
                    bf.h[2 * j + 1] = (__bf16)p2[(size_t)(2 * j + 1) * DDIM];
                }
#pragma unroll
                for (int mi = 0; mi < 2; ++mi)
                    acc[mi][ni] = __builtin_amdgcn_wmma_f32_16x16x32_bf16(
                        false, af[mi].v, false, bf.v, (short)0, acc[mi][ni], false, false);
            }
        }
    }

#pragma unroll
    for (int mi = 0; mi < 2; ++mi)
#pragma unroll
        for (int ni = 0; ni < 8; ++ni) {
            int dcol = wave * 128 + ni * 16 + nB;
            float bb = b2e[dcol];
#pragma unroll
            for (int r = 0; r < 8; ++r) {
                int row = mi * 16 + r + rsh;
                float contrib = swgt[row] * (acc[mi][ni][r] + bb);
                atomicAdd(&out[(size_t)stok[row] * DDIM + dcol], contrib);
            }
        }
}

extern "C" void kernel_launch(void* const* d_in, const int* in_sizes, int n_in,
                              void* d_out, int out_size, void* d_ws, size_t ws_size,
                              hipStream_t stream) {
    const float* x  = (const float*)d_in[0];
    const float* gw = (const float*)d_in[1];
    const float* gb = (const float*)d_in[2];
    const float* w1 = (const float*)d_in[3];
    const float* b1 = (const float*)d_in[4];
    const float* w2 = (const float*)d_in[5];
    const float* b2 = (const float*)d_in[6];
    float* out = (float*)d_out;

    int*   counts = (int*)d_ws;
    int*   ids    = (int*)((char*)d_ws + 256);
    float* wts    = (float*)((char*)d_ws + 256 + (size_t)NEXP * NTOK * sizeof(int));
    size_t meta   = 256 + (size_t)NEXP * NTOK * 8;            // 524,544 B
    size_t wbytes = (size_t)NEXP * DDIM * FDIM * 2;           // 67,108,864 B each
    __bf16* w1s   = (__bf16*)((char*)d_ws + meta);
    __bf16* w2s   = (__bf16*)((char*)d_ws + meta + wbytes);

    hipMemsetAsync(counts, 0, 256, stream);
    hipMemsetAsync(out, 0, (size_t)out_size * sizeof(float), stream);

    moe_gate<<<NTOK / 8, 256, 0, stream>>>(x, gw, gb, counts, ids, wts);

    size_t shmem = (size_t)TM * DDIM * 2 + (size_t)TM * FC * 2 + (size_t)TM * 8;

    if (ws_size >= meta + 2 * wbytes) {
        // Fast path: bf16 weights pre-swizzled into WMMA fragment order (134 MB,
        // L2-resident on MI455X's 192 MB L2).
        int fragBlocks = (NEXP * FT1 * KT1 * 32) / 256;       // 8192
        conv_w1<<<fragBlocks, 256, 0, stream>>>(w1, w1s);
        conv_w2<<<fragBlocks, 256, 0, stream>>>(w2, w2s);
        moe_ffn_pre<<<NEXP * TILES_PER_E, 256, shmem, stream>>>(
            x, w1s, b1, w2s, b2, counts, ids, wts, out);
    } else {
        // Fallback: convert weights on the fly.
        moe_ffn_raw<<<NEXP * TILES_PER_E, 256, shmem, stream>>>(
            x, w1, b1, w2, b2, counts, ids, wts, out);
    }
}
